// CrossAttentionLayer_52802327937147
// MI455X (gfx1250) — matrix operand order repaired and checked by reading, not compiled, not run on hardware
//
#include <hip/hip_runtime.h>
#include <hip/hip_bf16.h>
#include <stdint.h>

#define N1 8192
#define N2 8192
#define D1 256
#define D2 128

#define LDK 144   // padded LDS row (elements) for 32x128 row-major chunks (288B, 16B-aligned, conflict-free)
#define LDT 56    // padded LDS row (elements) for transposed chunks / P staging (112B)

#ifndef __has_builtin
#define __has_builtin(x) 0
#endif

#if __has_builtin(__builtin_amdgcn_global_load_async_to_lds_b128)
#define USE_ASYNC 1
#else
#define USE_ASYNC 0
#endif

#if USE_ASYNC
#if __has_builtin(__builtin_amdgcn_s_wait_asynccnt)
#define WAIT_ASYNC() __builtin_amdgcn_s_wait_asynccnt(0)
#else
#define WAIT_ASYNC() asm volatile("s_wait_asynccnt 0x0" ::: "memory")
#endif
#define AS1 __attribute__((address_space(1)))
#define AS3 __attribute__((address_space(3)))
typedef int gi32x4 __attribute__((vector_size(4 * sizeof(int))));
static __device__ __forceinline__ void async_cp16(const void* g, void* l) {
  __builtin_amdgcn_global_load_async_to_lds_b128(
      (AS1 gi32x4*)(uintptr_t)g, (AS3 gi32x4*)(uint32_t)(uintptr_t)l, 0, 0);
}
#endif

typedef __attribute__((ext_vector_type(16))) __bf16   v16bf;
typedef __attribute__((ext_vector_type(8)))  float    v8f;
typedef __attribute__((ext_vector_type(4)))  uint32_t u32x4;

struct B256 { u32x4 lo, hi; };

static __device__ __forceinline__ __bf16 f2bf(float f) {
  uint32_t u = __builtin_bit_cast(uint32_t, f);
  u += 0x7FFFu + ((u >> 16) & 1u);           // round-to-nearest-even
  uint16_t h = (uint16_t)(u >> 16);
  return __builtin_bit_cast(__bf16, h);
}

// A operand, 16x32 bf16 tile from row-major M[.. x ld] at (row0, k0).
// lane<16: row (lane&15), K = {0..7, 16..23}; lane>=16: same row, K = {8..15, 24..31}.
static __device__ __forceinline__ v16bf load_a16x32(const __bf16* M, int ld,
                                                    int row0, int k0, int lane) {
  const __bf16* p = M + (size_t)(row0 + (lane & 15)) * ld + k0 + ((lane >> 4) << 3);
  B256 t;
  t.lo = *(const u32x4*)(p);
  t.hi = *(const u32x4*)(p + 16);
  return __builtin_bit_cast(v16bf, t);
}

// B operand, 32x16 bf16 tile where B[k][n] = Y[row0+n][k0+k], Y row-major.
// lane n (n<16): K = {0..7, 16..23} of column n; lane n+16: K = {8..15, 24..31}.
static __device__ __forceinline__ v16bf load_bT32x16(const __bf16* Y, int ld,
                                                     int row0, int k0, int lane) {
  const __bf16* p = Y + (size_t)(row0 + (lane & 15)) * ld + k0 + ((lane >> 4) << 3);
  B256 t;
  t.lo = *(const u32x4*)(p);
  t.hi = *(const u32x4*)(p + 16);
  return __builtin_bit_cast(v16bf, t);
}

static __device__ __forceinline__ v8f wmma_bf16(v16bf a, v16bf b, v8f c) {
  return __builtin_amdgcn_wmma_f32_16x16x32_bf16(false, a, false, b,
                                                 (short)0, c, false, false);
}

// ---------------------------------------------------------------------------
// Chunk address helpers (128-thread blocks).
// Row-major chunk: 32 rows x 128 cols bf16 (512 x 16B), 4 chunks / thread.
// Transposed chunk (pass1): 128 rows x 32 cols (512 x 16B), 4 chunks / thread.
// Transposed chunk (pass2): 256 rows x 32 cols (1024 x 16B), 8 chunks / thread.
// ---------------------------------------------------------------------------
static __device__ __forceinline__ const __bf16* rm_src(const __bf16* src, int row_base, int c) {
  return src + (size_t)(row_base + (c >> 4)) * D2 + (c & 15) * 8;
}
static __device__ __forceinline__ __bf16* rm_dst(__bf16* s, int c) {
  return s + (c >> 4) * LDK + (c & 15) * 8;
}
static __device__ __forceinline__ const __bf16* tr_src(const __bf16* srcT, int ldg, int col_base, int c) {
  return srcT + (size_t)(c >> 2) * ldg + col_base + (c & 3) * 8;
}
static __device__ __forceinline__ __bf16* tr_dst(__bf16* s, int c) {
  return s + (c >> 2) * LDT + (c & 3) * 8;
}

#if USE_ASYNC
static __device__ __forceinline__ void stage_rm32x128(const __bf16* src, int row_base,
                                                      __bf16* s, int tid) {
#pragma unroll
  for (int j = 0; j < 4; ++j) {
    int c = tid + 128 * j;
    async_cp16(rm_src(src, row_base, c), rm_dst(s, c));
  }
}
static __device__ __forceinline__ void stage_tr128x32(const __bf16* srcT, int col_base,
                                                      __bf16* s, int tid) {
#pragma unroll
  for (int j = 0; j < 4; ++j) {
    int c = tid + 128 * j;
    async_cp16(tr_src(srcT, N2, col_base, c), tr_dst(s, c));
  }
}
static __device__ __forceinline__ void stage_tr256x32(const __bf16* srcT, int col_base,
                                                      __bf16* s, int tid) {
#pragma unroll
  for (int j = 0; j < 8; ++j) {
    int c = tid + 128 * j;
    async_cp16(tr_src(srcT, N1, col_base, c), tr_dst(s, c));
  }
}
#else
static __device__ __forceinline__ void ld_rm32x128(const __bf16* src, int row_base,
                                                   int tid, u32x4 a[4]) {
#pragma unroll
  for (int j = 0; j < 4; ++j) a[j] = *(const u32x4*)rm_src(src, row_base, tid + 128 * j);
}
static __device__ __forceinline__ void st_rm32x128(__bf16* s, int tid, const u32x4 a[4]) {
#pragma unroll
  for (int j = 0; j < 4; ++j) *(u32x4*)rm_dst(s, tid + 128 * j) = a[j];
}
static __device__ __forceinline__ void ld_tr128x32(const __bf16* srcT, int col_base,
                                                   int tid, u32x4 b[4]) {
#pragma unroll
  for (int j = 0; j < 4; ++j) b[j] = *(const u32x4*)tr_src(srcT, N2, col_base, tid + 128 * j);
}
static __device__ __forceinline__ void st_tr128x32(__bf16* s, int tid, const u32x4 b[4]) {
#pragma unroll
  for (int j = 0; j < 4; ++j) *(u32x4*)tr_dst(s, tid + 128 * j) = b[j];
}
static __device__ __forceinline__ void ld_tr256x32(const __bf16* srcT, int col_base,
                                                   int tid, u32x4 b[8]) {
#pragma unroll
  for (int j = 0; j < 8; ++j) b[j] = *(const u32x4*)tr_src(srcT, N1, col_base, tid + 128 * j);
}
static __device__ __forceinline__ void st_tr256x32(__bf16* s, int tid, const u32x4 b[8]) {
#pragma unroll
  for (int j = 0; j < 8; ++j) *(u32x4*)tr_dst(s, tid + 128 * j) = b[j];
}
#endif

// ---------------------------------------------------------------------------
// K0: f32 -> bf16 convert, optional transposed copy
// ---------------------------------------------------------------------------
__global__ void convert_bf16_kernel(const float* __restrict__ src,
                                    __bf16* __restrict__ dstB,
                                    __bf16* __restrict__ dstT, int R, int C) {
  int idx = blockIdx.x * blockDim.x + threadIdx.x;
  if (idx >= R * C) return;
  int r = idx / C, c = idx - r * C;
  __bf16 v = f2bf(src[idx]);
  dstB[idx] = v;
  if (dstT) dstT[(size_t)c * R + r] = v;
}

// ---------------------------------------------------------------------------
// K1: q = x1 @ W1^T + b1  -> qb (bf16 [N1 x D2]); one wave = 16 rows
// ---------------------------------------------------------------------------
__global__ __launch_bounds__(256) void qproj_kernel(const __bf16* __restrict__ x1b,
                                                    const __bf16* __restrict__ w1b,
                                                    const float*  __restrict__ b1,
                                                    __bf16* __restrict__ qb) {
  const int lane = threadIdx.x & 31;
  const int wave = threadIdx.x >> 5;
  const int row0 = (blockIdx.x * 8 + wave) * 16;

  v8f acc[8];
#pragma unroll
  for (int n = 0; n < 8; ++n) acc[n] = {};

#pragma unroll
  for (int kc = 0; kc < 8; ++kc) {
    v16bf a = load_a16x32(x1b, D1, row0, kc * 32, lane);
#pragma unroll
    for (int n = 0; n < 8; ++n) {
      v16bf b = load_bT32x16(w1b, D1, n * 16, kc * 32, lane);  // W1 rows = W1^T cols
      acc[n] = wmma_bf16(a, b, acc[n]);
    }
  }
  const int colw = lane & 15;
  const int rb   = (lane >> 4) << 3;
#pragma unroll
  for (int n = 0; n < 8; ++n) {
    float bias = b1[n * 16 + colw];
#pragma unroll
    for (int r = 0; r < 8; ++r)
      qb[(size_t)(row0 + rb + r) * D2 + n * 16 + colw] = f2bf(acc[n][r] + bias);
  }
}

// ---------------------------------------------------------------------------
// K2: flash-attention forward. Block = 4 waves, 64 query rows; key chunks of
// 32 double-buffered through LDS (async DMA when available).
// ---------------------------------------------------------------------------
__global__ __launch_bounds__(128) void attn_fwd_kernel(const __bf16* __restrict__ qb,
                                                       const __bf16* __restrict__ x2b,
                                                       const __bf16* __restrict__ x2t,
                                                       float* __restrict__ out1,
                                                       float* __restrict__ mg,
                                                       float* __restrict__ lg) {
  __shared__ __align__(16) __bf16 sX2[2][32 * LDK];    // keys x D2 (row-major)
  __shared__ __align__(16) __bf16 sX2T[2][D2 * LDT];   // D2 x 32 keys (transposed)
  __shared__ __align__(16) __bf16 sP[4][16 * LDT];     // per-wave P staging

  const int tid  = threadIdx.x;
  const int lane = tid & 31;
  const int wave = tid >> 5;
  const int row0 = (blockIdx.x * 4 + wave) * 16;
  __bf16* myP = &sP[wave][0];

  v16bf qA[4];
#pragma unroll
  for (int j = 0; j < 4; ++j) qA[j] = load_a16x32(qb, D2, row0, j * 32, lane);

  v8f O[8];
#pragma unroll
  for (int i = 0; i < 8; ++i) O[i] = {};
  v8f m, l = {};
#pragma unroll
  for (int r = 0; r < 8; ++r) m[r] = -3.0e38f;

  const int colw = lane & 15;
  const int rb   = (lane >> 4) << 3;

  // prologue: stage chunk 0
#if USE_ASYNC
  stage_rm32x128(x2b, 0, &sX2[0][0], tid);
  stage_tr128x32(x2t, 0, &sX2T[0][0], tid);
  WAIT_ASYNC();
#else
  {
    u32x4 a[4], b[4];
    ld_rm32x128(x2b, 0, tid, a);
    ld_tr128x32(x2t, 0, tid, b);
    st_rm32x128(&sX2[0][0], tid, a);
    st_tr128x32(&sX2T[0][0], tid, b);
  }
#endif
  __syncthreads();

  for (int kt = 0; kt < N2; kt += 32) {
    const int cur = (kt >> 5) & 1;
    const int nxt = cur ^ 1;
    const bool more = (kt + 32) < N2;

#if USE_ASYNC
    if (more) {                       // fire-and-forget DMA for next chunk
      stage_rm32x128(x2b, kt + 32, &sX2[nxt][0], tid);
      stage_tr128x32(x2t, kt + 32, &sX2T[nxt][0], tid);
    }
#else
    u32x4 nA[4], nB[4];
    if (more) {                       // issue next chunk's global loads early
      ld_rm32x128(x2b, kt + 32, tid, nA);
      ld_tr128x32(x2t, kt + 32, tid, nB);
    }
#endif

    const __bf16* cX2  = &sX2[cur][0];
    const __bf16* cX2T = &sX2T[cur][0];

    // S tiles: [16 queries x 16 keys], keys kt.. / kt+16..
    v8f S0 = {}, S1 = {};
#pragma unroll
    for (int j = 0; j < 4; ++j) {
      S0 = wmma_bf16(qA[j], load_bT32x16(cX2, LDK, 0,  j * 32, lane), S0);
      S1 = wmma_bf16(qA[j], load_bT32x16(cX2, LDK, 16, j * 32, lane), S1);
    }
    // online softmax: row max / sum via 16-lane xor shuffles
    v8f mn, scale, P0, P1;
#pragma unroll
    for (int r = 0; r < 8; ++r) {
      float t = fmaxf(S0[r], S1[r]);
#pragma unroll
      for (int s = 1; s < 16; s <<= 1) t = fmaxf(t, __shfl_xor(t, s, 32));
      mn[r]    = fmaxf(m[r], t);
      scale[r] = __expf(m[r] - mn[r]);
      P0[r]    = __expf(S0[r] - mn[r]);
      P1[r]    = __expf(S1[r] - mn[r]);
    }
#pragma unroll
    for (int r = 0; r < 8; ++r) {
      float t = P0[r] + P1[r];
#pragma unroll
      for (int s = 1; s < 16; s <<= 1) t += __shfl_xor(t, s, 32);
      l[r] = l[r] * scale[r] + t;
      m[r] = mn[r];
    }
#pragma unroll
    for (int i = 0; i < 8; ++i) O[i] *= scale;

    // stage P (bf16 16x32) through per-wave LDS -> A layout
#pragma unroll
    for (int r = 0; r < 8; ++r) {
      myP[(rb + r) * LDT + colw]      = f2bf(P0[r]);
      myP[(rb + r) * LDT + 16 + colw] = f2bf(P1[r]);
    }
    v16bf pA = load_a16x32(myP, LDT, 0, 0, lane);
    // O += P @ x2[kt:kt+32, :]   (B columns = rows of staged x2t chunk)
#pragma unroll
    for (int i = 0; i < 8; ++i)
      O[i] = wmma_bf16(pA, load_bT32x16(cX2T, LDT, i * 16, 0, lane), O[i]);

#if USE_ASYNC
    WAIT_ASYNC();
#else
    if (more) {                       // publish next chunk
      st_rm32x128(&sX2[nxt][0], tid, nA);
      st_tr128x32(&sX2T[nxt][0], tid, nB);
    }
#endif
    __syncthreads();
  }

  v8f rl;
#pragma unroll
  for (int r = 0; r < 8; ++r) rl[r] = 1.0f / l[r];
#pragma unroll
  for (int i = 0; i < 8; ++i)
#pragma unroll
    for (int r = 0; r < 8; ++r)
      out1[(size_t)(row0 + rb + r) * D2 + i * 16 + colw] = O[i][r] * rl[r];
  if (colw == 0) {
#pragma unroll
    for (int r = 0; r < 8; ++r) {
      mg[row0 + rb + r] = m[r];
      lg[row0 + rb + r] = l[r];
    }
  }
}

// ---------------------------------------------------------------------------
// K3: fused_x2 = attn^T @ x1. Block = 4 waves, 64 key rows; query chunks of
// 32 double-buffered through LDS. Recomputes S^T = x2 @ q^T, scales with m,l.
// ---------------------------------------------------------------------------
__global__ __launch_bounds__(128) void attn_bwd_kernel(const __bf16* __restrict__ qb,
                                                       const __bf16* __restrict__ x2b,
                                                       const __bf16* __restrict__ x1t,
                                                       const float* __restrict__ mg,
                                                       const float* __restrict__ lg,
                                                       float* __restrict__ out2) {
  __shared__ __align__(16) __bf16 sQ[2][32 * LDK];     // queries x D2 (row-major)
  __shared__ __align__(16) __bf16 sX1T[2][D1 * LDT];   // D1 x 32 queries (transposed)
  __shared__ __align__(16) __bf16 sP[4][16 * LDT];

  const int tid  = threadIdx.x;
  const int lane = tid & 31;
  const int wave = tid >> 5;
  const int j0   = (blockIdx.x * 4 + wave) * 16;
  __bf16* myP = &sP[wave][0];

  v16bf kA[4];
#pragma unroll
  for (int j = 0; j < 4; ++j) kA[j] = load_a16x32(x2b, D2, j0, j * 32, lane);

  v8f O2[16];
#pragma unroll
  for (int i = 0; i < 16; ++i) O2[i] = {};

  const int colw = lane & 15;
  const int rb   = (lane >> 4) << 3;

#if USE_ASYNC
  stage_rm32x128(qb, 0, &sQ[0][0], tid);
  stage_tr256x32(x1t, 0, &sX1T[0][0], tid);
  WAIT_ASYNC();
#else
  {
    u32x4 a[4], b[8];
    ld_rm32x128(qb, 0, tid, a);
    ld_tr256x32(x1t, 0, tid, b);
    st_rm32x128(&sQ[0][0], tid, a);
    st_tr256x32(&sX1T[0][0], tid, b);
  }
#endif
  __syncthreads();

  for (int it = 0; it < N1; it += 32) {
    const int cur = (it >> 5) & 1;
    const int nxt = cur ^ 1;
    const bool more = (it + 32) < N1;

#if USE_ASYNC
    if (more) {
      stage_rm32x128(qb, it + 32, &sQ[nxt][0], tid);
      stage_tr256x32(x1t, it + 32, &sX1T[nxt][0], tid);
    }
#else
    u32x4 nA[4], nB[8];
    if (more) {
      ld_rm32x128(qb, it + 32, tid, nA);
      ld_tr256x32(x1t, it + 32, tid, nB);
    }
#endif

    const __bf16* cQ   = &sQ[cur][0];
    const __bf16* cX1T = &sX1T[cur][0];

    float m0  = mg[it + colw];
    float m1  = mg[it + 16 + colw];
    float rl0 = 1.0f / lg[it + colw];
    float rl1 = 1.0f / lg[it + 16 + colw];

    // S^T tiles: [16 keys x 16 queries]
    v8f T0 = {}, T1 = {};
#pragma unroll
    for (int j = 0; j < 4; ++j) {
      T0 = wmma_bf16(kA[j], load_bT32x16(cQ, LDK, 0,  j * 32, lane), T0);
      T1 = wmma_bf16(kA[j], load_bT32x16(cQ, LDK, 16, j * 32, lane), T1);
    }
    // P^T = exp(S^T - m_col) / l_col, staged to LDS (16 keys x 32 queries)
#pragma unroll
    for (int r = 0; r < 8; ++r) {
      myP[(rb + r) * LDT + colw]      = f2bf(__expf(T0[r] - m0) * rl0);
      myP[(rb + r) * LDT + 16 + colw] = f2bf(__expf(T1[r] - m1) * rl1);
    }
    v16bf pA = load_a16x32(myP, LDT, 0, 0, lane);
    // O2 += P^T @ x1[it:it+32, :]   (B columns = rows of staged x1t chunk)
#pragma unroll
    for (int i = 0; i < 16; ++i)
      O2[i] = wmma_bf16(pA, load_bT32x16(cX1T, LDT, i * 16, 0, lane), O2[i]);

#if USE_ASYNC
    WAIT_ASYNC();
#else
    if (more) {
      st_rm32x128(&sQ[nxt][0], tid, nA);
      st_tr256x32(&sX1T[nxt][0], tid, nB);
    }
#endif
    __syncthreads();
  }
#pragma unroll
  for (int i = 0; i < 16; ++i)
#pragma unroll
    for (int r = 0; r < 8; ++r)
      out2[(size_t)(j0 + rb + r) * D1 + i * 16 + colw] = O2[i][r];
}

// ---------------------------------------------------------------------------
extern "C" void kernel_launch(void* const* d_in, const int* in_sizes, int n_in,
                              void* d_out, int out_size, void* d_ws, size_t ws_size,
                              hipStream_t stream) {
  const float* x1 = (const float*)d_in[0];   // [N1, D1]
  const float* x2 = (const float*)d_in[1];   // [N2, D2]
  const float* W1 = (const float*)d_in[2];   // [D2, D1]
  const float* b1 = (const float*)d_in[3];   // [D2]

  float* out1 = (float*)d_out;                    // fused_x1 [N1, D2]
  float* out2 = (float*)d_out + (size_t)N1 * D2;  // fused_x2 [N2, D1]

  char* ws = (char*)d_ws;
  size_t off = 0;
  __bf16* x1b = (__bf16*)(ws + off); off += (size_t)N1 * D1 * 2;   // 4 MB
  __bf16* x1t = (__bf16*)(ws + off); off += (size_t)N1 * D1 * 2;   // 4 MB
  __bf16* x2b = (__bf16*)(ws + off); off += (size_t)N2 * D2 * 2;   // 2 MB
  __bf16* x2t = (__bf16*)(ws + off); off += (size_t)N2 * D2 * 2;   // 2 MB
  __bf16* w1b = (__bf16*)(ws + off); off += (size_t)D2 * D1 * 2;   // 64 KB
  __bf16* qb  = (__bf16*)(ws + off); off += (size_t)N1 * D2 * 2;   // 2 MB
  float*  mg  = (float*)(ws + off);  off += (size_t)N1 * 4;
  float*  lg  = (float*)(ws + off);  off += (size_t)N1 * 4;

  convert_bf16_kernel<<<(N1 * D1 + 255) / 256, 256, 0, stream>>>(x1, x1b, x1t, N1, D1);
  convert_bf16_kernel<<<(N2 * D2 + 255) / 256, 256, 0, stream>>>(x2, x2b, x2t, N2, D2);
  convert_bf16_kernel<<<(D2 * D1 + 255) / 256, 256, 0, stream>>>(W1, w1b, nullptr, D2, D1);

  qproj_kernel<<<N1 / (16 * 8), 256, 0, stream>>>(x1b, w1b, b1, qb);
  attn_fwd_kernel<<<N1 / (16 * 4), 128, 0, stream>>>(qb, x2b, x2t, out1, mg, lg);
  attn_bwd_kernel<<<N2 / (16 * 4), 128, 0, stream>>>(qb, x2b, x1t, mg, lg, out2);
}
